// Net_37830071943759
// MI455X (gfx1250) — matrix operand order repaired and checked by reading, not compiled, not run on hardware
//
#include <hip/hip_runtime.h>
#include <hip/hip_bf16.h>

typedef __bf16 bf16;
typedef __attribute__((ext_vector_type(16))) __bf16 v16bf;
typedef __attribute__((ext_vector_type(8)))  __bf16 v8bf;   // 16 bytes
typedef __attribute__((ext_vector_type(8)))  float  v8f;

union frag16 { v16bf v; v8bf h[2]; };

// ---------------------------------------------------------------------------
// WMMA fragment helpers (wave32, CDNA5 ISA 7.12.2 layouts)
// ---------------------------------------------------------------------------
__device__ __forceinline__ v8f wmma_bf16(v16bf a, v16bf b, v8f c) {
    return __builtin_amdgcn_wmma_f32_16x16x32_bf16(false, a, false, b, (short)0, c,
                                                   false, false);
}

// A 16x32 bf16, row-major source with leading dim ld.
__device__ __forceinline__ v16bf load_a16(const bf16* __restrict__ A, int ld, int row0,
                                          int k0) {
    const int lane = threadIdx.x & 31;
    const bf16* p  = A + (size_t)(row0 + (lane & 15)) * ld + k0 + ((lane >> 4) << 3);
    frag16 u;
    u.h[0] = *(const v8bf*)(p);
    u.h[1] = *(const v8bf*)(p + 16);
    return u.v;
}

// B fragment from packed (fragment-major) storage: frag f, lane L -> 32 contiguous B
__device__ __forceinline__ v16bf load_b_packed(const bf16* __restrict__ Bp, int f) {
    const int lane = threadIdx.x & 31;
    const bf16* p  = Bp + ((size_t)f * 32 + lane) * 16;
    frag16 u;
    u.h[0] = *(const v8bf*)(p);
    u.h[1] = *(const v8bf*)(p + 8);
    return u.v;
}

// ---------------------------------------------------------------------------
// Repack fp32 weight [K x 128] row-major -> bf16 WMMA-fragment-major.
// ---------------------------------------------------------------------------
__global__ void repack_b_kernel(const float* __restrict__ W, bf16* __restrict__ Bp,
                                size_t total /* K*128 */) {
    size_t t = (size_t)blockIdx.x * blockDim.x + threadIdx.x;
    if (t >= total) return;
    int e    = (int)(t & 15);
    int lane = (int)((t >> 4) & 31);
    int f    = (int)(t >> 9);
    int kblk = f >> 3;
    int nt   = f & 7;
    int k    = kblk * 32 + ((lane >> 4) << 3) + (e & 7) + ((e >> 3) << 4);
    int n    = nt * 16 + (lane & 15);
    Bp[t] = (bf16)W[(size_t)k * 128 + n];
}

// ---------------------------------------------------------------------------
// Utility kernels
// ---------------------------------------------------------------------------
__global__ void cvt_bf16_kernel(const float* __restrict__ s, bf16* __restrict__ d,
                                size_t n) {
    size_t i = (size_t)blockIdx.x * blockDim.x + threadIdx.x;
    if (i < n) d[i] = (bf16)s[i];
}

__global__ void fill_f32_kernel(float* __restrict__ p, float v, size_t n) {
    size_t i = (size_t)blockIdx.x * blockDim.x + threadIdx.x;
    if (i < n) p[i] = v;
}

__global__ void deg_accum_kernel(const int* __restrict__ dst, float* __restrict__ deg,
                                 int E) {
    int e = blockIdx.x * blockDim.x + threadIdx.x;
    if (e < E) atomicAdd(&deg[dst[e]], 1.0f);
}

__global__ void rsqrt_kernel(const float* __restrict__ deg, float* __restrict__ dinv,
                             size_t n) {
    size_t i = (size_t)blockIdx.x * blockDim.x + threadIdx.x;
    if (i < n) dinv[i] = rsqrtf(deg[i]);
}

__global__ void acc_init_kernel(const float* __restrict__ xl,
                                const float* __restrict__ dinvI,
                                const float* __restrict__ dinvD,
                                float* __restrict__ accI, float* __restrict__ accD,
                                size_t total) {
    size_t i = (size_t)blockIdx.x * blockDim.x + threadIdx.x;
    if (i >= total) return;
    int v   = (int)(i >> 7);
    float x = xl[i];
    float a = dinvI[v];
    float b = dinvD[v];
    accI[i] = a * a * x;
    accD[i] = b * b * x;
}

__global__ void scatter_kernel(const int* __restrict__ src, const int* __restrict__ dst,
                               const float* __restrict__ dinv,
                               const float* __restrict__ xl, float* __restrict__ acc,
                               int E) {
    long long t = (long long)blockIdx.x * blockDim.x + threadIdx.x;
    int e = (int)(t >> 5);
    if (e >= E) return;
    int h  = ((int)t & 31) * 4;
    int s  = src[e];
    int d  = dst[e];
    float norm = dinv[s] * dinv[d];
    const float4 v = *(const float4*)(xl + (size_t)s * 128 + h);
    float* a = acc + (size_t)d * 128 + h;
    atomicAdd(a + 0, norm * v.x);
    atomicAdd(a + 1, norm * v.y);
    atomicAdd(a + 2, norm * v.z);
    atomicAdd(a + 3, norm * v.w);
}

__global__ void finalize_z_kernel(const float* __restrict__ accI,
                                  const float* __restrict__ accD,
                                  const float* __restrict__ b, bf16* __restrict__ zb,
                                  size_t total) {
    size_t i = (size_t)blockIdx.x * blockDim.x + threadIdx.x;
    if (i >= total) return;
    int h   = (int)(i & 127);
    float u = accI[i] + b[h]; u = u > 0.f ? u : 0.f;
    float w = accD[i] + b[h]; w = w > 0.f ? w : 0.f;
    zb[i] = (bf16)(u + w);
}

// ---------------------------------------------------------------------------
// xl = x @ W_conv : one wave -> TWO 16-row tiles x 128 cols; B staged in LDS.
// __launch_bounds__(256,1): real block size is 256, so the allocator may use
// the full VGPR budget (2x8 accumulator tiles + frags ~ 170 VGPRs, no spills).
// ---------------------------------------------------------------------------
__global__ void __launch_bounds__(256, 1)
gemm_conv_kernel(const bf16* __restrict__ xb, const bf16* __restrict__ Bp,
                 float* __restrict__ xl, int Nrows) {
    extern __shared__ char smem[];
    bf16* bs = (bf16*)smem;                       // 256*128 bf16 = 64 KB
    const size_t nchunks = (size_t)256 * 128 / 8; // 4096 x 16B
    for (size_t c = threadIdx.x; c < nchunks; c += blockDim.x)
        ((v8bf*)bs)[c] = ((const v8bf*)Bp)[c];
    __syncthreads();

    int wave = (int)(((size_t)blockIdx.x * blockDim.x + threadIdx.x) >> 5);
    int row0 = wave * 32;
    if (row0 >= Nrows) return;
    const bool t1   = (row0 + 16) < Nrows;        // tail guard (N % 32 == 16)
    const int  row1 = t1 ? row0 + 16 : row0;
    const int lane  = threadIdx.x & 31;

    v8f acc0[8], acc1[8];
#pragma unroll
    for (int nt = 0; nt < 8; ++nt) {
        acc0[nt] = (v8f){0.f, 0.f, 0.f, 0.f, 0.f, 0.f, 0.f, 0.f};
        acc1[nt] = (v8f){0.f, 0.f, 0.f, 0.f, 0.f, 0.f, 0.f, 0.f};
    }

#pragma unroll
    for (int ks = 0; ks < 8; ++ks) {              // K = 256
        v16bf a0 = load_a16(xb, 256, row0, ks * 32);
        v16bf a1 = load_a16(xb, 256, row1, ks * 32);
#pragma unroll
        for (int nt = 0; nt < 8; ++nt) {
            v16bf b  = load_b_packed(bs, ks * 8 + nt);
            acc0[nt] = wmma_bf16(a0, b, acc0[nt]);
            acc1[nt] = wmma_bf16(a1, b, acc1[nt]);
        }
    }
    const int nlo = lane & 15;
    const int mh  = (lane >> 4) * 8;
#pragma unroll
    for (int nt = 0; nt < 8; ++nt)
#pragma unroll
        for (int r = 0; r < 8; ++r) {
            xl[(size_t)(row0 + mh + r) * 128 + nt * 16 + nlo] = acc0[nt][r];
            if (t1)
                xl[(size_t)(row1 + mh + r) * 128 + nt * 16 + nlo] = acc1[nt][r];
        }
}

// ---------------------------------------------------------------------------
// Fused decoder. Segment id of a K-block is k0>>7 (uniform & compile-time
// under full unroll), so the [zi | zj | |zi-zj|] gather branches fold away.
// One wave -> TWO 16-pair tiles.
// ---------------------------------------------------------------------------
__device__ __forceinline__ v16bf load_pair_a(const bf16* __restrict__ zb, int i, int j,
                                             int k0, int lane) {
    const int seg  = k0 >> 7;                     // 0: zi, 1: zj, 2: |zi-zj|
    const int cb   = (k0 & 127) + ((lane >> 4) << 3);
    const bf16* zi = zb + (size_t)i * 128;
    const bf16* zj = zb + (size_t)j * 128;
    frag16 u;
#pragma unroll
    for (int run = 0; run < 2; ++run) {
        const int c = cb + run * 16;
        v8bf r;
        if (seg == 0)      r = *(const v8bf*)(zi + c);
        else if (seg == 1) r = *(const v8bf*)(zj + c);
        else {
            v8bf a = *(const v8bf*)(zi + c);
            v8bf b = *(const v8bf*)(zj + c);
#pragma unroll
            for (int e = 0; e < 8; ++e) {
                float d = (float)a[e] - (float)b[e];
                r[e] = (bf16)(d < 0.f ? -d : d);
            }
        }
        u.h[run] = r;
    }
    return u.v;
}

__device__ __forceinline__ void decode_epilogue(const v8f* acc, const float* b1,
                                                const float* W2, float b2v,
                                                float* __restrict__ out, int row0,
                                                int lane) {
    const int nlo = lane & 15;
    float s[8];
#pragma unroll
    for (int r = 0; r < 8; ++r) s[r] = 0.f;
#pragma unroll
    for (int nt = 0; nt < 8; ++nt) {
        int n    = nt * 16 + nlo;
        float w2 = W2[n];
        float bb = b1[n];
#pragma unroll
        for (int r = 0; r < 8; ++r) {
            float h = acc[nt][r] + bb;
            h = h > 0.f ? h : 0.f;
            s[r] += h * w2;
        }
    }
#pragma unroll
    for (int off = 1; off < 16; off <<= 1)
#pragma unroll
        for (int r = 0; r < 8; ++r) s[r] += __shfl_xor(s[r], off, 16);

    if (nlo == 0) {
        const int base = row0 + (lane >> 4) * 8;
        float4 lo = {s[0] + b2v, s[1] + b2v, s[2] + b2v, s[3] + b2v};
        float4 hi = {s[4] + b2v, s[5] + b2v, s[6] + b2v, s[7] + b2v};
        *(float4*)(out + base)     = lo;
        *(float4*)(out + base + 4) = hi;
    }
}

__global__ void __launch_bounds__(256, 1)
decode_kernel(const bf16* __restrict__ zb, const int* __restrict__ ei,
              const int* __restrict__ ej, const bf16* __restrict__ Bp1,
              const float* __restrict__ b1, const float* __restrict__ W2,
              const float* __restrict__ b2, float* __restrict__ out, int P) {
    extern __shared__ char smem[];
    bf16* bs = (bf16*)smem;                       // 384*128 bf16 = 96 KB
    const size_t nchunks = (size_t)384 * 128 / 8; // 6144 x 16B
    for (size_t c = threadIdx.x; c < nchunks; c += blockDim.x)
        ((v8bf*)bs)[c] = ((const v8bf*)Bp1)[c];
    __syncthreads();

    int wave = (int)(((size_t)blockIdx.x * blockDim.x + threadIdx.x) >> 5);
    int row0 = wave * 32;
    if (row0 >= P) return;
    const bool t1 = (row0 + 16) < P;              // P % 32 == 0 in practice
    const int lane = threadIdx.x & 31;
    const int m    = lane & 15;
    const int p0   = row0 + m;
    const int p1   = t1 ? row0 + 16 + m : p0;
    const int i0 = ei[p0], j0 = ej[p0];
    const int i1 = ei[p1], j1 = ej[p1];

    v8f acc0[8], acc1[8];
#pragma unroll
    for (int nt = 0; nt < 8; ++nt) {
        acc0[nt] = (v8f){0.f, 0.f, 0.f, 0.f, 0.f, 0.f, 0.f, 0.f};
        acc1[nt] = (v8f){0.f, 0.f, 0.f, 0.f, 0.f, 0.f, 0.f, 0.f};
    }

#pragma unroll
    for (int ks = 0; ks < 12; ++ks) {             // K = 384
        v16bf a0 = load_pair_a(zb, i0, j0, ks * 32, lane);
        v16bf a1 = load_pair_a(zb, i1, j1, ks * 32, lane);
#pragma unroll
        for (int nt = 0; nt < 8; ++nt) {
            v16bf b  = load_b_packed(bs, ks * 8 + nt);
            acc0[nt] = wmma_bf16(a0, b, acc0[nt]);
            acc1[nt] = wmma_bf16(a1, b, acc1[nt]);
        }
    }

    float b2v = b2[0];
    decode_epilogue(acc0, b1, W2, b2v, out, row0, lane);
    if (t1) decode_epilogue(acc1, b1, W2, b2v, out, row0 + 16, lane);
}

// ---------------------------------------------------------------------------
extern "C" void kernel_launch(void* const* d_in, const int* in_sizes, int n_in,
                              void* d_out, int out_size, void* d_ws, size_t ws_size,
                              hipStream_t stream) {
    const float* x   = (const float*)d_in[0];
    const int*   eI  = (const int*)d_in[1];
    const int*   eD  = (const int*)d_in[2];
    const int*   eL  = (const int*)d_in[3];
    const float* Wc  = (const float*)d_in[4];
    const float* bc  = (const float*)d_in[5];
    const float* W1  = (const float*)d_in[6];
    const float* b1  = (const float*)d_in[7];
    const float* W2  = (const float*)d_in[8];
    const float* b2  = (const float*)d_in[9];
    float*       out = (float*)d_out;

    const int F = 256, H = 128;
    const int N = in_sizes[0] / F;       // 50000
    const int E = in_sizes[1] / 2;       // 800000
    const int P = in_sizes[3] / 2;       // 500000

    // ---- workspace layout (256B aligned) ----
    char* ws = (char*)d_ws;
    size_t off = 0;
    auto take = [&](size_t bytes) {
        size_t r = off;
        off = (off + bytes + 255) & ~(size_t)255;
        return r;
    };
    size_t o_xb   = take((size_t)N * F * sizeof(bf16));     // aliased by accI later
    size_t o_Wcp  = take((size_t)F * H * sizeof(bf16));     // packed conv weights
    size_t o_W1p  = take((size_t)3 * H * H * sizeof(bf16)); // packed MLP weights
    size_t o_xl   = take((size_t)N * H * sizeof(float));
    size_t o_deg  = take((size_t)2 * N * sizeof(float));
    size_t o_dinv = take((size_t)2 * N * sizeof(float));
    size_t o_accD = take((size_t)N * H * sizeof(float));
    size_t o_zb   = take((size_t)N * H * sizeof(bf16));
    (void)ws_size;

    bf16*  xb    = (bf16*)(ws + o_xb);
    bf16*  Wcp   = (bf16*)(ws + o_Wcp);
    bf16*  W1p   = (bf16*)(ws + o_W1p);
    float* xl    = (float*)(ws + o_xl);
    float* degI  = (float*)(ws + o_deg);
    float* degD  = degI + N;
    float* dinvI = (float*)(ws + o_dinv);
    float* dinvD = dinvI + N;
    float* accI  = (float*)(ws + o_xb);   // alias: xb dead after gemm_conv
    float* accD  = (float*)(ws + o_accD);
    bf16*  zb    = (bf16*)(ws + o_zb);

    const int BT = 256;
    auto blocks = [&](size_t n) { return (unsigned)((n + BT - 1) / BT); };

    // 1) conversions / weight repack
    cvt_bf16_kernel<<<blocks((size_t)N * F), BT, 0, stream>>>(x, xb, (size_t)N * F);
    repack_b_kernel<<<blocks((size_t)F * H), BT, 0, stream>>>(Wc, Wcp, (size_t)F * H);
    repack_b_kernel<<<blocks((size_t)3 * H * H), BT, 0, stream>>>(W1, W1p,
                                                                  (size_t)3 * H * H);

    // 2) degrees (init 1.0 for self-loops)
    fill_f32_kernel<<<blocks((size_t)2 * N), BT, 0, stream>>>(degI, 1.0f, (size_t)2 * N);
    deg_accum_kernel<<<blocks(E), BT, 0, stream>>>(eI + E, degI, E);
    deg_accum_kernel<<<blocks(E), BT, 0, stream>>>(eD + E, degD, E);

    // 3) xl = x @ W_conv  (WMMA, 32 rows/wave, B in LDS)
    {
        size_t waves = (size_t)(N + 31) / 32;
        gemm_conv_kernel<<<blocks(waves * 32), BT, 65536, stream>>>(xb, Wcp, xl, N);
    }

    // 4) dinv = rsqrt(deg)
    rsqrt_kernel<<<blocks((size_t)2 * N), BT, 0, stream>>>(degI, dinvI, (size_t)2 * N);

    // 5) self-loop init (fully overwrites accI alias of xb)
    acc_init_kernel<<<blocks((size_t)N * H), BT, 0, stream>>>(xl, dinvI, dinvD, accI,
                                                              accD, (size_t)N * H);

    // 6) edge scatter for both graphs
    scatter_kernel<<<blocks((size_t)E * 32), BT, 0, stream>>>(eI, eI + E, dinvI, xl,
                                                              accI, E);
    scatter_kernel<<<blocks((size_t)E * 32), BT, 0, stream>>>(eD, eD + E, dinvD, xl,
                                                              accD, E);

    // 7) z = relu(accI + bc) + relu(accD + bc) -> bf16
    finalize_z_kernel<<<blocks((size_t)N * H), BT, 0, stream>>>(accI, accD, bc, zb,
                                                                (size_t)N * H);

    // 8) fused decoder MLP (WMMA, 32 pairs/wave, W1 in LDS) -> out[P]
    {
        size_t waves = (size_t)(P + 31) / 32;
        decode_kernel<<<blocks(waves * 32), BT, 98304, stream>>>(zb, eL, eL + P, W1p, b1,
                                                                 W2, b2, out, P);
    }
    (void)n_in;
    (void)out_size;
}
